// CustomMultiHeadAttention_59571196395803
// MI455X (gfx1250) — compile-verified, matrix operand order checked by reading, not run on hardware
//
#include <hip/hip_runtime.h>
#include <hip/hip_bf16.h>
#include <math.h>

// ---------------- Types ----------------
typedef __bf16 bf16_t;
typedef bf16_t v16bf __attribute__((ext_vector_type(16)));
typedef float  v8f   __attribute__((ext_vector_type(8)));

union FragU {
    v16bf v;
    unsigned int d[8];
    unsigned short u[16];
};

#define T_SEQ 2048
#define BSZ   2
#define EMB   1024
#define NHEAD 16
#define HDIM  64
#define BH    (BSZ * NHEAD)      // 32
#define M1    (T_SEQ * BSZ)      // 4096 rows for in/out proj
#define N1    (3 * EMB)          // 3072
#define LOG2E 1.44269504088896f

// float -> bf16 (round to nearest even)
__device__ __forceinline__ unsigned short f2bf(float f) {
    unsigned int u = __float_as_uint(f);
    u += 0x7FFFu + ((u >> 16) & 1u);
    return (unsigned short)(u >> 16);
}

// Load a 16x32 bf16 fragment (A-layout; also used for B' = row-major-over-K).
// Lane L holds row (L&15). Lanes 0-15: K in {0..7, 16..23}; lanes 16-31: K in {8..15, 24..31}.
// All quantities in 16-bit element units; (row*strideH + kOff) must be even.
__device__ __forceinline__ v16bf load_frag16x32(const unsigned short* base, int row,
                                                int strideH, int kOff) {
    int lane = threadIdx.x & 31;
    int klo  = (lane & 16) ? 8 : 0;
    const unsigned int* p =
        (const unsigned int*)(base + (size_t)row * (size_t)strideH + kOff + klo);
    FragU f;
#pragma unroll
    for (int j = 0; j < 8; ++j) {
        int dw = (j & 3) + ((j >> 2) << 3);   // dwords 0..3 then 8..11
        f.d[j] = p[dw];
    }
    return f.v;
}

__device__ __forceinline__ v8f wmma_bf16(v16bf a, v16bf b, v8f c) {
    return __builtin_amdgcn_wmma_f32_16x16x32_bf16(false, a, false, b,
                                                   (short)0, c, false, false);
}

__device__ __forceinline__ float rowmax16(float v) {
#pragma unroll
    for (int m = 1; m < 16; m <<= 1) v = fmaxf(v, __shfl_xor(v, m, 16));
    return v;
}
__device__ __forceinline__ float rowsum16(float v) {
#pragma unroll
    for (int m = 1; m < 16; m <<= 1) v += __shfl_xor(v, m, 16);
    return v;
}

// ---------------- Kernel 1a: f32 -> bf16 ----------------
__global__ void cvt_f32_bf16(const float* __restrict__ src,
                             unsigned short* __restrict__ dst, int n) {
    int i = blockIdx.x * blockDim.x + threadIdx.x;
    if (i < n) dst[i] = f2bf(src[i]);
}

// ---------------- Kernel 1b: transpose + convert: dst[n*K+k] = src[k*N+n] ----------------
__global__ void cvt_transpose(const float* __restrict__ src,
                              unsigned short* __restrict__ dst, int K, int N) {
    int i = blockIdx.x * blockDim.x + threadIdx.x;
    if (i < K * N) {
        int k = i & (K - 1);     // K is a power of two (1024)
        int n = i >> 10;
        dst[i] = f2bf(src[(size_t)k * N + n]);
    }
}

// ---------------- shared GEMM mainloop (ping-pong, no register rotation) ----------------
// A: rowA = m0+cbase, stride EMB. B': rows n0+t*16+cbase, stride EMB.
// acc[4] accumulates a 16x64 tile over K=EMB.
__device__ __forceinline__ void gemm_mainloop(const unsigned short* __restrict__ A,
                                              const unsigned short* __restrict__ Bt,
                                              int m0, int n0, int cbase, v8f acc[4]) {
    int rowA = m0 + cbase;

    v16bf aA = load_frag16x32(A, rowA, EMB, 0);
    v16bf bA[4];
#pragma unroll
    for (int t = 0; t < 4; ++t)
        bA[t] = load_frag16x32(Bt, n0 + t * 16 + cbase, EMB, 0);

    for (int kk = 0; kk < EMB; kk += 64) {
        // set B: fragments for kk+32 (always in range: EMB/32 is even)
        v16bf aB = load_frag16x32(A, rowA, EMB, kk + 32);
        v16bf bB[4];
#pragma unroll
        for (int t = 0; t < 4; ++t)
            bB[t] = load_frag16x32(Bt, n0 + t * 16 + cbase, EMB, kk + 32);
        __builtin_prefetch(A + (size_t)rowA * EMB + kk + 96, 0, 3);
        __builtin_prefetch(Bt + (size_t)(n0 + cbase) * EMB + kk + 96, 0, 3);

        // consume set A (loaded a half-iteration ago)
#pragma unroll
        for (int t = 0; t < 4; ++t) acc[t] = wmma_bf16(aA, bA[t], acc[t]);

        // reload set A for kk+64 (clamped at tail; branch-free, extra load harmless)
        int k2 = (kk + 64 < EMB) ? (kk + 64) : 0;
        aA = load_frag16x32(A, rowA, EMB, k2);
#pragma unroll
        for (int t = 0; t < 4; ++t)
            bA[t] = load_frag16x32(Bt, n0 + t * 16 + cbase, EMB, k2);

        // consume set B
#pragma unroll
        for (int t = 0; t < 4; ++t) acc[t] = wmma_bf16(aB, bB[t], acc[t]);
    }
}

// ---------------- Kernel 2: in-projection GEMM + QKV scatter ----------------
__global__ __launch_bounds__(256) void gemm_qkv(
        const unsigned short* __restrict__ xb,
        const unsigned short* __restrict__ w1t,
        const float* __restrict__ bias,
        unsigned short* __restrict__ qb,
        unsigned short* __restrict__ kb,
        unsigned short* __restrict__ vtb) {
    int wid   = threadIdx.x >> 5;
    int lane  = threadIdx.x & 31;
    int hi    = (lane >> 4) & 1;
    int cbase = lane & 15;
    int m0    = blockIdx.y * 128 + wid * 16;
    int n0    = blockIdx.x * 64;

    v8f acc[4];
#pragma unroll
    for (int t = 0; t < 4; ++t)
#pragma unroll
        for (int i = 0; i < 8; ++i) acc[t][i] = 0.0f;

    gemm_mainloop(xb, w1t, m0, n0, cbase, acc);

#pragma unroll
    for (int t = 0; t < 4; ++t) {
#pragma unroll
        for (int i = 0; i < 8; ++i) {
            int mrow = m0 + i + 8 * hi;
            int f    = n0 + t * 16 + cbase;
            float val = acc[t][i] + bias[f];
            int which = f >> 10;          // 0:q 1:k 2:v
            int e     = f & 1023;
            int h     = e >> 6;
            int d     = e & 63;
            int tseq  = mrow >> 1;        // row = t*B + b, B=2
            int b     = mrow & 1;
            int bh    = b * NHEAD + h;
            if (which == 0) {
                qb[((size_t)(bh * T_SEQ + tseq)) * HDIM + d] = f2bf(val * 0.125f);
            } else if (which == 1) {
                kb[((size_t)(bh * T_SEQ + tseq)) * HDIM + d] = f2bf(val);
            } else {
                vtb[((size_t)(bh * HDIM + d)) * T_SEQ + tseq] = f2bf(val);
            }
        }
    }
}

// ---------------- attention step: one 32-key block ----------------
__device__ __forceinline__ void attn_step(v16bf qa0, v16bf qa1,
                                          const v16bf kf[4],
                                          const unsigned short* __restrict__ vtb,
                                          int j0, float mrun[8], float lrun[8],
                                          v8f acc[4], unsigned short* pl,
                                          int hi, int cbase) {
    // V fragments for this block (consumed after softmax -> latency hidden)
    v16bf vf[4];
#pragma unroll
    for (int t = 0; t < 4; ++t)
        vf[t] = load_frag16x32(vtb, t * 16 + cbase, T_SEQ, j0);

    // ---- S = Q * K^T for 32 keys ----
    v8f s0, s1;
#pragma unroll
    for (int i = 0; i < 8; ++i) { s0[i] = 0.0f; s1[i] = 0.0f; }
    s0 = wmma_bf16(qa0, kf[0], s0);
    s0 = wmma_bf16(qa1, kf[1], s0);
    s1 = wmma_bf16(qa0, kf[2], s1);
    s1 = wmma_bf16(qa1, kf[3], s1);

    // ---- online softmax (rows live on 16-lane halves) ----
    float alpha[8];
#pragma unroll
    for (int i = 0; i < 8; ++i) {
        float mx = rowmax16(fmaxf(s0[i], s1[i]));
        float mn = fmaxf(mrun[i], mx);
        float sc = exp2f((mrun[i] - mn) * LOG2E);
        float p0 = exp2f((s0[i] - mn) * LOG2E);
        float p1 = exp2f((s1[i] - mn) * LOG2E);
        lrun[i]  = lrun[i] * sc + rowsum16(p0 + p1);
        mrun[i]  = mn;
        alpha[i] = sc;
        s0[i] = p0;
        s1[i] = p1;
    }
#pragma unroll
    for (int t = 0; t < 4; ++t)
#pragma unroll
        for (int i = 0; i < 8; ++i) acc[t][i] *= alpha[i];

    // ---- P: C-layout -> A-layout via padded LDS ----
#pragma unroll
    for (int i = 0; i < 8; ++i) {
        int rr = i + 8 * hi;
        pl[rr * 34 + cbase]      = f2bf(s0[i]);
        pl[rr * 34 + 16 + cbase] = f2bf(s1[i]);
    }
    asm volatile("s_wait_dscnt 0" ::: "memory");
    v16bf pa = load_frag16x32(pl, cbase, 34, 0);

    // ---- acc += P * V ----
#pragma unroll
    for (int t = 0; t < 4; ++t) acc[t] = wmma_bf16(pa, vf[t], acc[t]);
}

// ---------------- Kernel 3: flash attention ----------------
// Grid (T/64, BH); 128 threads = 4 waves; each wave: 16 queries x full key loop.
// K fragments ping-pong across pairs of 32-key blocks: no register rotation.
__global__ __launch_bounds__(128) void attn_kernel(
        const unsigned short* __restrict__ qg,
        const unsigned short* __restrict__ kg,
        const unsigned short* __restrict__ vtg,
        unsigned short* __restrict__ aog) {
    __shared__ unsigned short Plds[4][16 * 34];

    int wid   = threadIdx.x >> 5;
    int lane  = threadIdx.x & 31;
    int hi    = (lane >> 4) & 1;
    int cbase = lane & 15;
    int bh    = blockIdx.y;
    int q0    = blockIdx.x * 64 + wid * 16;

    const unsigned short* qb  = qg  + (size_t)bh * T_SEQ * HDIM;
    const unsigned short* kb  = kg  + (size_t)bh * T_SEQ * HDIM;
    const unsigned short* vtb = vtg + (size_t)bh * HDIM * T_SEQ;

    v16bf qa0 = load_frag16x32(qb, q0 + cbase, HDIM, 0);
    v16bf qa1 = load_frag16x32(qb, q0 + cbase, HDIM, 32);

    float mrun[8], lrun[8];
    v8f acc[4];
#pragma unroll
    for (int i = 0; i < 8; ++i) { mrun[i] = -INFINITY; lrun[i] = 0.0f; }
#pragma unroll
    for (int t = 0; t < 4; ++t)
#pragma unroll
        for (int i = 0; i < 8; ++i) acc[t][i] = 0.0f;

    unsigned short* pl = Plds[wid];

    // preload K fragments for block 0: index = tile*2 + khalf
    v16bf kfA[4], kfB[4];
#pragma unroll
    for (int t = 0; t < 2; ++t)
#pragma unroll
        for (int h = 0; h < 2; ++h)
            kfA[t * 2 + h] = load_frag16x32(kb, t * 16 + cbase, HDIM, h * 32);

    for (int j0 = 0; j0 < T_SEQ; j0 += 64) {
        // prefetch K for block j0+32 while processing block j0
#pragma unroll
        for (int t = 0; t < 2; ++t)
#pragma unroll
            for (int h = 0; h < 2; ++h)
                kfB[t * 2 + h] =
                    load_frag16x32(kb, j0 + 32 + t * 16 + cbase, HDIM, h * 32);

        attn_step(qa0, qa1, kfA, vtb, j0, mrun, lrun, acc, pl, hi, cbase);

        // prefetch K for block j0+64 (clamped at tail) while processing j0+32
        int jn = (j0 + 64 < T_SEQ) ? (j0 + 64) : 0;
#pragma unroll
        for (int t = 0; t < 2; ++t)
#pragma unroll
            for (int h = 0; h < 2; ++h)
                kfA[t * 2 + h] =
                    load_frag16x32(kb, jn + t * 16 + cbase, HDIM, h * 32);

        attn_step(qa0, qa1, kfB, vtb, j0 + 32, mrun, lrun, acc, pl, hi, cbase);
    }

    // ---- normalize and store (flat (BH,T,D) == faithful reshape to (B,T,E)) ----
    float linv[8];
#pragma unroll
    for (int i = 0; i < 8; ++i) linv[i] = 1.0f / lrun[i];
#pragma unroll
    for (int t = 0; t < 4; ++t) {
#pragma unroll
        for (int i = 0; i < 8; ++i) {
            float o = acc[t][i] * linv[i];
            int trow = q0 + i + 8 * hi;
            aog[((size_t)(bh * T_SEQ + trow)) * HDIM + t * 16 + cbase] = f2bf(o);
        }
    }
}

// ---------------- Kernel 4: out-projection GEMM (fp32 out + bias) ----------------
__global__ __launch_bounds__(256) void gemm_out(
        const unsigned short* __restrict__ aob,
        const unsigned short* __restrict__ w2t,
        const float* __restrict__ outb,
        float* __restrict__ out) {
    int wid   = threadIdx.x >> 5;
    int lane  = threadIdx.x & 31;
    int hi    = (lane >> 4) & 1;
    int cbase = lane & 15;
    int m0    = blockIdx.y * 128 + wid * 16;
    int n0    = blockIdx.x * 64;

    v8f acc[4];
#pragma unroll
    for (int t = 0; t < 4; ++t)
#pragma unroll
        for (int i = 0; i < 8; ++i) acc[t][i] = 0.0f;

    gemm_mainloop(aob, w2t, m0, n0, cbase, acc);

#pragma unroll
    for (int t = 0; t < 4; ++t) {
#pragma unroll
        for (int i = 0; i < 8; ++i) {
            int mrow = m0 + i + 8 * hi;
            int col  = n0 + t * 16 + cbase;
            out[(size_t)mrow * EMB + col] = acc[t][i] + outb[col];
        }
    }
}

// ---------------- Host launch ----------------
extern "C" void kernel_launch(void* const* d_in, const int* in_sizes, int n_in,
                              void* d_out, int out_size, void* d_ws, size_t ws_size,
                              hipStream_t stream) {
    const float* x       = (const float*)d_in[0];   // (T,B,E)
    const float* w1      = (const float*)d_in[1];   // (E,3E)
    const float* b1      = (const float*)d_in[2];   // (3E,)
    const float* w2      = (const float*)d_in[3];   // (E,E)
    const float* b2      = (const float*)d_in[4];   // (E,)
    float*       out     = (float*)d_out;

    unsigned short* xb  = (unsigned short*)d_ws;           // 4194304 halves
    unsigned short* w1t = xb  + (size_t)M1 * EMB;          // 3145728
    unsigned short* w2t = w1t + (size_t)N1 * EMB;          // 1048576
    unsigned short* qb  = w2t + (size_t)EMB * EMB;         // 4194304
    unsigned short* kb  = qb  + (size_t)BH * T_SEQ * HDIM;
    unsigned short* vtb = kb  + (size_t)BH * T_SEQ * HDIM;
    unsigned short* aob = vtb + (size_t)BH * T_SEQ * HDIM;

    // 1) conversions
    {
        int n = M1 * EMB;
        cvt_f32_bf16<<<(n + 255) / 256, 256, 0, stream>>>(x, xb, n);
    }
    {
        int n = EMB * N1;
        cvt_transpose<<<(n + 255) / 256, 256, 0, stream>>>(w1, w1t, EMB, N1);
    }
    {
        int n = EMB * EMB;
        cvt_transpose<<<(n + 255) / 256, 256, 0, stream>>>(w2, w2t, EMB, EMB);
    }

    // 2) in-projection + QKV scatter
    gemm_qkv<<<dim3(N1 / 64, M1 / 128), 256, 0, stream>>>(xb, w1t, b1, qb, kb, vtb);

    // 3) attention
    attn_kernel<<<dim3(T_SEQ / 64, BH), 128, 0, stream>>>(qb, kb, vtb, aob);

    // 4) out projection
    gemm_out<<<dim3(EMB / 64, M1 / 128), 256, 0, stream>>>(aob, w2t, b2, out);
}
